// QuantizedLinear_17343077941843
// MI455X (gfx1250) — matrix operand order repaired and checked by reading, not compiled, not run on hardware
//
#include <hip/hip_runtime.h>

// GEMM: out[M,N] = x[M,K] * W[N,K]^T * rowscale + bias, W is int4 packed 2 nibbles per int32.
// M = B*S = 4096, K = 4096, N = 11008.

typedef __attribute__((ext_vector_type(16))) __bf16 v16bf;
typedef __attribute__((ext_vector_type(8)))  float  v8f;

#define M_DIM 4096
#define K_DIM 4096
#define N_DIM 11008

#define BM 128
#define BN 128
#define BK 64
#define LDA 72  // padded bf16 stride (144B) to spread LDS banks
#define LDB 72

__global__ __launch_bounds__(256)
void q4_gemm_bf16_wmma(const float* __restrict__ x,
                       const int*   __restrict__ pw,
                       const float* __restrict__ scales,
                       const float* __restrict__ bias,
                       float* __restrict__ out)
{
    __shared__ __bf16 sA[BM * LDA];
    __shared__ __bf16 sB[BN * LDB];

    const int tid    = threadIdx.x;
    const int lane   = tid & 31;
    const int wave   = tid >> 5;
    const int wave_n = wave & 3;   // 4 waves across N (32 cols each)
    const int wave_m = wave >> 2;  // 2 waves across M (64 rows each)

    const int n0 = blockIdx.x * BN;
    const int m0 = blockIdx.y * BM;

    const int hgrp = lane >> 4;    // 0: lanes 0-15, 1: lanes 16-31
    const int l16  = lane & 15;

    v8f acc[4][2];
#pragma unroll
    for (int mi = 0; mi < 4; ++mi)
#pragma unroll
        for (int ni = 0; ni < 2; ++ni)
            acc[mi][ni] = v8f{};

    // Staging coordinates (constant across k-loop)
    const int a_row = tid >> 4;          // + r*16, r = 0..7   (16 thr/row * 4 floats = 64 K)
    const int a_kq  = (tid & 15) * 4;    // 4 floats
    const int b_row = tid >> 3;          // + r*32, r = 0..3   (8 thr/row * 4 words  = 32 words)
    const int b_wq  = (tid & 7) * 4;     // 4 int32 words (= 8 weights)

    for (int k0 = 0; k0 < K_DIM; k0 += BK) {
        // ---- stage A: x tile fp32 -> bf16 (native cvt, lets clang emit packed cvt) ----
#pragma unroll
        for (int r = 0; r < 8; ++r) {
            const int row = r * 16 + a_row;
            const float4 xv = *(const float4*)(x + (size_t)(m0 + row) * K_DIM + k0 + a_kq);
            union { __bf16 h[4]; uint2 u; } pk;
            pk.h[0] = (__bf16)xv.x; pk.h[1] = (__bf16)xv.y;
            pk.h[2] = (__bf16)xv.z; pk.h[3] = (__bf16)xv.w;
            *(uint2*)(&sA[row * LDA + a_kq]) = pk.u;
        }
        // ---- stage B: unpack int4 -> (q-8) as bf16 (exact) ----
#pragma unroll
        for (int r = 0; r < 4; ++r) {
            const int row = r * 32 + b_row;
            const int4 wv = *(const int4*)(pw + (size_t)(n0 + row) * (K_DIM / 2) + (k0 >> 1) + b_wq);
            const int w4[4] = { wv.x, wv.y, wv.z, wv.w };
            union { __bf16 h[8]; uint4 u; } pk;
#pragma unroll
            for (int j = 0; j < 4; ++j) {
                pk.h[2 * j + 0] = (__bf16)(float)((w4[j] & 0xF) - 8);
                pk.h[2 * j + 1] = (__bf16)(float)(((w4[j] >> 4) & 0xF) - 8);
            }
            *(uint4*)(&sB[row * LDB + 2 * b_wq]) = pk.u;
        }

        // prefetch next k-tile while this one is consumed
        if (k0 + BK < K_DIM) {
            __builtin_prefetch(x + (size_t)(m0 + a_row) * K_DIM + (k0 + BK) + a_kq, 0, 0);
            __builtin_prefetch(pw + (size_t)(n0 + b_row) * (K_DIM / 2) + ((k0 + BK) >> 1) + b_wq, 0, 0);
        }

        __syncthreads();

        // ---- two K=32 sub-steps: 16 WMMAs per wave between barriers ----
#pragma unroll
        for (int ks = 0; ks < 2; ++ks) {
            const int kb = ks * 32;

            // A (16x32 bf16): lane<16 -> M=l16, K {0..7, 16..23}; lane>=16 -> K {8..15, 24..31}
            v16bf afr[4];
#pragma unroll
            for (int mi = 0; mi < 4; ++mi) {
                const int m = wave_m * 64 + mi * 16 + l16;
                const __bf16* src = &sA[m * LDA + kb + hgrp * 8];
                union { v16bf v; uint4 u[2]; } f;
                f.u[0] = *(const uint4*)(src);        // K base + 0..7
                f.u[1] = *(const uint4*)(src + 16);   // K base + 16..23
                afr[mi] = f.v;
            }
            // B (32x16 bf16): lane -> N=l16, K {hgrp*8 + 0..7, 16 + hgrp*8 + 0..7} (N-major)
            v16bf bfr[2];
#pragma unroll
            for (int ni = 0; ni < 2; ++ni) {
                const int n = wave_n * 32 + ni * 16 + l16;
                const __bf16* src = &sB[n * LDB + kb + hgrp * 8];
                union { v16bf v; uint4 u[2]; } f;
                f.u[0] = *(const uint4*)(src);
                f.u[1] = *(const uint4*)(src + 16);
                bfr[ni] = f.v;
            }

#pragma unroll
            for (int mi = 0; mi < 4; ++mi)
#pragma unroll
                for (int ni = 0; ni < 2; ++ni)
                    acc[mi][ni] = __builtin_amdgcn_wmma_f32_16x16x32_bf16(
                        false, afr[mi], false, bfr[ni],
                        (short)0, acc[mi][ni], false, false);
        }

        __syncthreads();
    }

    // ---- epilogue: y = acc * scale[n] + bias[n], fp32 out ----
    // C layout: VGPR r, lane L -> m = 8*(L>=16)+r, n = L&15
#pragma unroll
    for (int ni = 0; ni < 2; ++ni) {
        const int n = n0 + wave_n * 32 + ni * 16 + l16;
        const float s  = scales[n];
        const float bz = bias[n];
#pragma unroll
        for (int mi = 0; mi < 4; ++mi) {
            const int mbase = m0 + wave_m * 64 + mi * 16 + hgrp * 8;
#pragma unroll
            for (int r = 0; r < 8; ++r) {
                out[(size_t)(mbase + r) * N_DIM + n] = acc[mi][ni][r] * s + bz;
            }
        }
    }
}

extern "C" void kernel_launch(void* const* d_in, const int* in_sizes, int n_in,
                              void* d_out, int out_size, void* d_ws, size_t ws_size,
                              hipStream_t stream) {
    const float* x      = (const float*)d_in[0];
    const int*   pw     = (const int*)  d_in[1];
    const float* scales = (const float*)d_in[2];
    const float* bias   = (const float*)d_in[3];
    float*       out    = (float*)d_out;

    dim3 grid(N_DIM / BN, M_DIM / BM);   // 86 x 32
    dim3 block(256);
    q4_gemm_bf16_wmma<<<grid, block, 0, stream>>>(x, pw, scales, bias, out);
}
